// StanceDetectionModel_67662914781719
// MI455X (gfx1250) — matrix operand order repaired and checked by reading, not compiled, not run on hardware
//
#include <hip/hip_runtime.h>

// ---------------------------------------------------------------------------
// StanceDetectionModel for MI455X (gfx1250, wave32, WMMA bf16)
//   V=50000 E=300 H=512 C=3 B=64 T_TEXT=256 T_TOPIC=16
// Strategy:
//   * x-projections of both GRU gates precomputed as big bf16 WMMA GEMMs
//     (each wave: 16x64 C strip; explicit ping-pong double buffering: WMMAs
//      consume one fragment set while the other set's loads are in flight;
//      no rotation movs)
//   * sequential h-recurrence in ONE 1024-thread workgroup, h in LDS (bf16),
//     ping-pong pipelined weight fragments across the 64 WMMA pairs per step
//   * attention softmax + context, fc1 via WMMA (+ReLU), scalar fc2
// ---------------------------------------------------------------------------

typedef __attribute__((ext_vector_type(16))) __bf16 v16bf;
typedef __attribute__((ext_vector_type(8)))  float  v8f;

union Frag {
    unsigned int u[8];
    v16bf        v;
};

__device__ __forceinline__ unsigned short f2bf(float f) {
    unsigned int u = __builtin_bit_cast(unsigned int, f);
    unsigned int r = u + 0x7FFFu + ((u >> 16) & 1u);   // round-to-nearest-even
    return (unsigned short)(r >> 16);
}
__device__ __forceinline__ float bf2f(unsigned short h) {
    unsigned int u = ((unsigned int)h) << 16;
    return __builtin_bit_cast(float, u);
}

// ---------------------------------------------------------------------------
// Pack fp32 weight slab W[row0 .. row0+Krows) x N (row-major, ld = N) into the
// per-lane WMMA B-fragment layout:
//   out[((ks*ntiles + nt)*32 + lane)*8 + v] = bf16pair( W[k][n], W[k+1][n] )
//   with k = ks*32 + (lane/16)*16 + v*2,  n = nt*16 + lane%16
// Zero-pads k >= Krows (handles E=300 -> Kp=320).
// ---------------------------------------------------------------------------
__global__ void pack_w_kernel(const float* __restrict__ W, int row0, int Krows,
                              int N, unsigned int* __restrict__ out,
                              int ksteps, int ntiles) {
    int i = blockIdx.x * 256 + threadIdx.x;
    int total = ksteps * ntiles * 256;
    if (i >= total) return;
    int v    = i & 7;
    int lane = (i >> 3) & 31;
    int nt   = (i >> 8) % ntiles;
    int ks   = i / (256 * ntiles);
    int k = ks * 32 + (lane >> 4) * 16 + v * 2;
    int n = nt * 16 + (lane & 15);
    float a = (k     < Krows) ? W[(long)(row0 + k)     * N + n] : 0.f;
    float b = (k + 1 < Krows) ? W[(long)(row0 + k + 1) * N + n] : 0.f;
    out[i] = (unsigned int)f2bf(a) | ((unsigned int)f2bf(b) << 16);
}

// Gather token embeddings -> bf16, row m = t*B + b, K padded 300 -> 320 (zeros)
__global__ void gather_embed_kernel(const int* __restrict__ tok,
                                    const float* __restrict__ emb,
                                    unsigned short* __restrict__ X,
                                    int Bn, int T, int E, int Kp) {
    long i = (long)blockIdx.x * 256 + threadIdx.x;
    long total = (long)T * Bn * Kp;
    if (i >= total) return;
    int  e = (int)(i % Kp);
    long m = i / Kp;
    int  t = (int)(m / Bn);
    int  b = (int)(m % Bn);
    float v = 0.f;
    if (e < E) v = emb[(long)tok[b * T + t] * E + e];
    X[i] = f2bf(v);
}

// ---------------------------------------------------------------------------
// Generic GEMM: C[M,N] = act( A[M,Kp](bf16,row-major) @ Bpack + bias )
// One wave per 16x64 C strip (4 consecutive n-tiles). Ping-pong double
// buffering: WMMAs of step ks overlap the loads of step ks+1.
// M multiple of 16, N multiple of 64, Kp multiple of 32.
// ---------------------------------------------------------------------------
__global__ void gemm_bf16_kernel(const unsigned short* __restrict__ A,
                                 const unsigned int* __restrict__ Bpack,
                                 const float* __restrict__ bias,
                                 float* __restrict__ C,
                                 int M, int N, int Kp, int act) {
    int wave = threadIdx.x >> 5;
    int lane = threadIdx.x & 31;
    int ntiles  = N >> 4;
    int ngroups = N >> 6;                     // groups of 4 n-tiles
    long wid  = (long)blockIdx.x * 8 + wave;
    long wtot = (long)(M >> 4) * ngroups;
    if (wid >= wtot) return;                  // wave-uniform: EXEC stays full
    int mt = (int)(wid / ngroups);
    int ng = (int)(wid % ngroups);
    int kh   = lane >> 4;
    int mrow = mt * 16 + (lane & 15);

    auto loadA = [&](int ks, Frag& a) {
        const unsigned short* arow = A + (long)mrow * Kp + ks * 32;
#pragma unroll
        for (int v = 0; v < 8; ++v) {
            int k = kh * 16 + v * 2;
            a.u[v] = *(const unsigned int*)(arow + k);
        }
    };
    auto loadB4 = [&](int ks, Frag* b) {
        const unsigned int* bq =
            Bpack + (((long)ks * ntiles + ng * 4) * 32 + lane) * 8;
#pragma unroll
        for (int j = 0; j < 4; ++j)
#pragma unroll
            for (int v = 0; v < 8; ++v) b[j].u[v] = bq[(long)j * 256 + v];
    };

    v8f acc[4] = {{}, {}, {}, {}};
    int ksteps = Kp >> 5;

    Frag a0, b0[4], a1, b1[4];                // ping-pong buffer sets
    loadA(0, a0);
    loadB4(0, b0);
#pragma unroll 1
    for (int ks = 0; ks < ksteps; ks += 2) {
        // phase 0: load set1 for ks+1, consume set0 (ks)
        if (ks + 1 < ksteps) {
            loadA(ks + 1, a1);
            loadB4(ks + 1, b1);
        }
        if (ks + 2 < ksteps)
            __builtin_prefetch(
                Bpack + (((long)(ks + 2) * ntiles + ng * 4) * 32 + lane) * 8,
                0, 3);
#pragma unroll
        for (int j = 0; j < 4; ++j)
            acc[j] = __builtin_amdgcn_wmma_f32_16x16x32_bf16(
                         false, a0.v, false, b0[j].v, (short)0, acc[j], false, false);
        // phase 1: load set0 for ks+2, consume set1 (ks+1)
        if (ks + 1 < ksteps) {
            if (ks + 2 < ksteps) {
                loadA(ks + 2, a0);
                loadB4(ks + 2, b0);
            }
#pragma unroll
            for (int j = 0; j < 4; ++j)
                acc[j] = __builtin_amdgcn_wmma_f32_16x16x32_bf16(
                             false, a1.v, false, b1[j].v, (short)0, acc[j], false, false);
        }
    }
#pragma unroll
    for (int j = 0; j < 4; ++j) {
        int n = (ng * 4 + j) * 16 + (lane & 15);
        float bi = bias[n];
#pragma unroll
        for (int r = 0; r < 8; ++r) {
            int m = mt * 16 + kh * 8 + r;
            float val = acc[j][r] + bi;
            if (act) val = fmaxf(val, 0.f);
            C[(long)m * N + n] = val;
        }
    }
}

// ---------------------------------------------------------------------------
// Sequential minGRU recurrence, single workgroup of 1024 threads (32 waves).
//   h: [64,512] bf16 in LDS.  Per step, per wave: 4 (m,n)-tiles x 2 gates,
//   64 WMMA pairs pipelined with ping-pong weight fragment sets (no movs).
//   z = sigmoid(XZ[t] + h@WzH), ht = tanh(XO[t] + h@WoH), h = (1-z)h + z ht
//   out[t*64+b][:] = h_new   (per-step h needed by attention)
// ---------------------------------------------------------------------------
__global__ __launch_bounds__(1024) void gru_seq_kernel(
        const unsigned int* __restrict__ WzH,   // packed [16][32][32][8]
        const unsigned int* __restrict__ WoH,
        const float* __restrict__ XZ,           // [T*64, 512]
        const float* __restrict__ XO,
        float* __restrict__ out,                // [T*64, 512]
        int T) {
    __shared__ unsigned short hbf[64 * 512];
    int tid = threadIdx.x;
    for (int i = tid; i < 64 * 512; i += 1024) hbf[i] = 0;
    __syncthreads();

    int wave = tid >> 5, lane = tid & 31;
    int mt  = wave & 3;        // 0..3  (batch tiles: 64/16)
    int nt0 = wave >> 2;       // 0..7  (n tiles nt0 + {0,8,16,24})
    int kh   = lane >> 4;
    int mrow = mt * 16 + (lane & 15);

    auto loadAh = [&](int ks, Frag& a) {
#pragma unroll
        for (int v = 0; v < 8; ++v) {
            int k = ks * 32 + kh * 16 + v * 2;
            a.u[v] = *(const unsigned int*)&hbf[mrow * 512 + k];
        }
    };
    auto loadW = [&](int ks, int j, Frag& bz, Frag& bo) {
        long off = (((long)ks * 32 + (nt0 + j * 8)) * 32 + lane) * 8;
#pragma unroll
        for (int v = 0; v < 8; ++v) bz.u[v] = WzH[off + v];
#pragma unroll
        for (int v = 0; v < 8; ++v) bo.u[v] = WoH[off + v];
    };

    for (int t = 0; t < T; ++t) {
        v8f accZ[4] = {{}, {}, {}, {}};
        v8f accO[4] = {{}, {}, {}, {}};

        Frag a_cur, bzA, boA, bzB, boB;       // ping-pong weight sets
        loadAh(0, a_cur);
        loadW(0, 0, bzA, boA);
        // iterate WMMA pairs two at a time: it0 = 2p (set A), it1 = 2p+1 (set B)
#pragma unroll 2
        for (int p = 0; p < 32; ++p) {
            int it0 = 2 * p;
            int it1 = it0 + 1;
            int it2 = it0 + 2;
            // load set B for it1 while consuming set A (it0)
            loadW(it1 >> 2, it1 & 3, bzB, boB);
            accZ[it0 & 3] = __builtin_amdgcn_wmma_f32_16x16x32_bf16(
                false, a_cur.v, false, bzA.v, (short)0, accZ[it0 & 3], false, false);
            accO[it0 & 3] = __builtin_amdgcn_wmma_f32_16x16x32_bf16(
                false, a_cur.v, false, boA.v, (short)0, accO[it0 & 3], false, false);
            // load set A for it2 while consuming set B (it1)
            if (it2 < 64) loadW(it2 >> 2, it2 & 3, bzA, boA);
            accZ[it1 & 3] = __builtin_amdgcn_wmma_f32_16x16x32_bf16(
                false, a_cur.v, false, bzB.v, (short)0, accZ[it1 & 3], false, false);
            accO[it1 & 3] = __builtin_amdgcn_wmma_f32_16x16x32_bf16(
                false, a_cur.v, false, boB.v, (short)0, accO[it1 & 3], false, false);
            // advance A-fragment (LDS, cheap) when finishing a K-step
            if ((it1 & 3) == 3 && it2 < 64) loadAh(it2 >> 2, a_cur);
        }

        long base = (long)t * 64 * 512;
        float hn[4][8];
#pragma unroll
        for (int j = 0; j < 4; ++j) {
#pragma unroll
            for (int r = 0; r < 8; ++r) {
                int m = mt * 16 + kh * 8 + r;
                int n = (nt0 + j * 8) * 16 + (lane & 15);
                long idx = base + (long)m * 512 + n;
                float z  = 1.f / (1.f + __expf(-(accZ[j][r] + XZ[idx])));
                float ht = tanhf(accO[j][r] + XO[idx]);
                float ho = bf2f(hbf[m * 512 + n]);
                float hv = (1.f - z) * ho + z * ht;
                hn[j][r] = hv;
                out[idx] = hv;
            }
        }
        __syncthreads();     // all waves finished reading old h
#pragma unroll
        for (int j = 0; j < 4; ++j) {
#pragma unroll
            for (int r = 0; r < 8; ++r) {
                int m = mt * 16 + kh * 8 + r;
                int n = (nt0 + j * 8) * 16 + (lane & 15);
                hbf[m * 512 + n] = f2bf(hn[j][r]);
            }
        }
        __syncthreads();     // new h visible before next step
    }
}

// ---------------------------------------------------------------------------
// Attention: scores[b,t] = text_out[t,b,:].attW[0:512] + topic_h[b,:].attW[512:]
//            + att_b ; softmax over t ; context[b,:] = sum_t w * text_out
// One block per batch row, 256 threads (= T).
// ---------------------------------------------------------------------------
__global__ void attention_kernel(const float* __restrict__ text_out,
                                 const float* __restrict__ topic_h,
                                 const float* __restrict__ attW,
                                 const float* __restrict__ attb,
                                 float* __restrict__ wout,     // [64,256]
                                 float* __restrict__ context)  // [64,512]
{
    __shared__ float sw[256];
    __shared__ float red[256];
    int b = blockIdx.x, t = threadIdx.x;

    float td = attW[512 + 2 * t]     * topic_h[b * 512 + 2 * t]
             + attW[512 + 2 * t + 1] * topic_h[b * 512 + 2 * t + 1];
    red[t] = td;
    __syncthreads();
    for (int s = 128; s > 0; s >>= 1) { if (t < s) red[t] += red[t + s]; __syncthreads(); }
    float topdot = red[0];
    __syncthreads();

    float score = topdot + attb[0];
    const float* row = text_out + ((long)t * 64 + b) * 512;
    for (int h = 0; h < 512; ++h) score += row[h] * attW[h];

    red[t] = score;
    __syncthreads();
    for (int s = 128; s > 0; s >>= 1) { if (t < s) red[t] = fmaxf(red[t], red[t + s]); __syncthreads(); }
    float mx = red[0];
    __syncthreads();

    float e = __expf(score - mx);
    red[t] = e;
    __syncthreads();
    for (int s = 128; s > 0; s >>= 1) { if (t < s) red[t] += red[t + s]; __syncthreads(); }
    float w = e / red[0];
    __syncthreads();

    wout[b * 256 + t] = w;
    sw[t] = w;
    __syncthreads();

    for (int c = t; c < 512; c += 256) {
        float acc = 0.f;
        for (int tt = 0; tt < 256; ++tt)
            acc += sw[tt] * text_out[((long)tt * 64 + b) * 512 + c];
        context[b * 512 + c] = acc;
    }
}

// feat = bf16([context, topic_h])  -> [64, 1024]
__global__ void featpack_kernel(const float* __restrict__ context,
                                const float* __restrict__ topic_h,
                                unsigned short* __restrict__ feat) {
    int i = blockIdx.x * 256 + threadIdx.x;   // 64*1024
    int b = i >> 10, j = i & 1023;
    float v = (j < 512) ? context[b * 512 + j] : topic_h[b * 512 + (j - 512)];
    feat[i] = f2bf(v);
}

// logits[b,c] = hid[b,:] . fc2_W[:,c] + fc2_b[c]   (64x512x3, fp32)
__global__ void fc2_kernel(const float* __restrict__ hid,
                           const float* __restrict__ W,
                           const float* __restrict__ bias,
                           float* __restrict__ out) {
    int i = threadIdx.x;
    if (i >= 192) return;
    int b = i / 3, c = i % 3;
    float acc = bias[c];
    for (int h = 0; h < 512; ++h) acc += hid[b * 512 + h] * W[h * 3 + c];
    out[i] = acc;
}

// ---------------------------------------------------------------------------
extern "C" void kernel_launch(void* const* d_in, const int* in_sizes, int n_in,
                              void* d_out, int out_size, void* d_ws, size_t ws_size,
                              hipStream_t stream) {
    (void)in_sizes; (void)n_in; (void)out_size; (void)ws_size;
    const int Bn = 64, Tt = 256, Tp = 16, E = 300, H = 512, EKP = 320;

    const int*   text     = (const int*)  d_in[0];
    const int*   topic    = (const int*)  d_in[1];
    const float* emb      = (const float*)d_in[2];
    const float* text_Wz  = (const float*)d_in[3];
    const float* text_bz  = (const float*)d_in[4];
    const float* text_Wo  = (const float*)d_in[5];
    const float* text_bo  = (const float*)d_in[6];
    const float* topic_Wz = (const float*)d_in[7];
    const float* topic_bz = (const float*)d_in[8];
    const float* topic_Wo = (const float*)d_in[9];
    const float* topic_bo = (const float*)d_in[10];
    const float* att_W    = (const float*)d_in[11];
    const float* att_b    = (const float*)d_in[12];
    const float* fc1_W    = (const float*)d_in[13];
    const float* fc1_b    = (const float*)d_in[14];
    const float* fc2_W    = (const float*)d_in[15];
    const float* fc2_b    = (const float*)d_in[16];
    float* out = (float*)d_out;             // [192 logits | 16384 weights]

    // ---- workspace bump allocator (~117 MB total) ----
    char* p = (char*)d_ws;
    auto alloc = [&](size_t sz) { void* r = (void*)p; p += (sz + 255) & ~(size_t)255; return r; };

    const int xk = EKP / 32, hk = H / 32, fk = 1024 / 32, nt = H / 16; // 10,16,32,32
    unsigned int* WzX  = (unsigned int*)alloc((size_t)xk * nt * 256 * 4);
    unsigned int* WoX  = (unsigned int*)alloc((size_t)xk * nt * 256 * 4);
    unsigned int* WzHt = (unsigned int*)alloc((size_t)hk * nt * 256 * 4);
    unsigned int* WoHt = (unsigned int*)alloc((size_t)hk * nt * 256 * 4);
    unsigned int* WzXp = (unsigned int*)alloc((size_t)xk * nt * 256 * 4);
    unsigned int* WoXp = (unsigned int*)alloc((size_t)xk * nt * 256 * 4);
    unsigned int* WzHp = (unsigned int*)alloc((size_t)hk * nt * 256 * 4);
    unsigned int* WoHp = (unsigned int*)alloc((size_t)hk * nt * 256 * 4);
    unsigned int* F1P  = (unsigned int*)alloc((size_t)fk * nt * 256 * 4);
    unsigned short* Xt   = (unsigned short*)alloc((size_t)Tt * Bn * EKP * 2);
    unsigned short* Xp   = (unsigned short*)alloc((size_t)Tp * Bn * EKP * 2);
    float* XZt  = (float*)alloc((size_t)Tt * Bn * H * 4);
    float* XOt  = (float*)alloc((size_t)Tt * Bn * H * 4);
    float* XZp  = (float*)alloc((size_t)Tp * Bn * H * 4);
    float* XOp  = (float*)alloc((size_t)Tp * Bn * H * 4);
    float* tout = (float*)alloc((size_t)Tt * Bn * H * 4);   // text_out [t*64+b, 512]
    float* pout = (float*)alloc((size_t)Tp * Bn * H * 4);
    float* ctx  = (float*)alloc((size_t)Bn * H * 4);
    unsigned short* feat = (unsigned short*)alloc((size_t)Bn * 1024 * 2);
    float* hid  = (float*)alloc((size_t)Bn * H * 4);
    const float* topic_h = pout + (size_t)(Tp - 1) * Bn * H;

    // ---- 1. repack weights into WMMA B-fragment layout ----
    auto packN = [&](const float* W, int row0, int Krows, int N, unsigned int* o, int ks) {
        int total = ks * (N / 16) * 256;
        pack_w_kernel<<<(total + 255) / 256, 256, 0, stream>>>(W, row0, Krows, N, o, ks, N / 16);
    };
    packN(text_Wz,  0, E,    H, WzX,  xk);
    packN(text_Wo,  0, E,    H, WoX,  xk);
    packN(text_Wz,  E, H,    H, WzHt, hk);
    packN(text_Wo,  E, H,    H, WoHt, hk);
    packN(topic_Wz, 0, E,    H, WzXp, xk);
    packN(topic_Wo, 0, E,    H, WoXp, xk);
    packN(topic_Wz, E, H,    H, WzHp, hk);
    packN(topic_Wo, E, H,    H, WoHp, hk);
    packN(fc1_W,    0, 1024, H, F1P,  fk);

    // ---- 2. embedding gather -> bf16 (K padded to 320) ----
    {
        long n = (long)Tt * Bn * EKP;
        gather_embed_kernel<<<(int)((n + 255) / 256), 256, 0, stream>>>(text, emb, Xt, Bn, Tt, E, EKP);
        n = (long)Tp * Bn * EKP;
        gather_embed_kernel<<<(int)((n + 255) / 256), 256, 0, stream>>>(topic, emb, Xp, Bn, Tp, E, EKP);
    }

    // ---- 3. x-projections: XZ/XO = X @ W[:E] + b   (bf16 WMMA GEMMs) ----
    auto gemm = [&](const unsigned short* A, const unsigned int* Bp, const float* bias,
                    float* Cm, int M, int N, int Kp, int act) {
        long waves = (long)(M / 16) * (N / 64);   // one wave per 16x64 strip
        gemm_bf16_kernel<<<(int)((waves + 7) / 8), 256, 0, stream>>>(A, Bp, bias, Cm, M, N, Kp, act);
    };
    gemm(Xt, WzX,  text_bz,  XZt, Tt * Bn, H, EKP, 0);
    gemm(Xt, WoX,  text_bo,  XOt, Tt * Bn, H, EKP, 0);
    gemm(Xp, WzXp, topic_bz, XZp, Tp * Bn, H, EKP, 0);
    gemm(Xp, WoXp, topic_bo, XOp, Tp * Bn, H, EKP, 0);

    // ---- 4. sequential minGRU scans (single-WGP persistent kernels) ----
    gru_seq_kernel<<<1, 1024, 0, stream>>>(WzHt, WoHt, XZt, XOt, tout, Tt);
    gru_seq_kernel<<<1, 1024, 0, stream>>>(WzHp, WoHp, XZp, XOp, pout, Tp);

    // ---- 5. attention: scores -> softmax (weights to d_out) -> context ----
    attention_kernel<<<Bn, 256, 0, stream>>>(tout, topic_h, att_W, att_b, out + 192, ctx);

    // ---- 6. fc1 (WMMA + ReLU) and fc2 (logits to d_out) ----
    featpack_kernel<<<(Bn * 1024) / 256, 256, 0, stream>>>(ctx, topic_h, feat);
    gemm(feat, F1P, fc1_b, hid, Bn, H, 1024, 1);
    fc2_kernel<<<1, 192, 0, stream>>>(hid, fc2_W, fc2_b, out);
}
